// MambaEncoder_57509612093794
// MI455X (gfx1250) — compile-verified, matrix operand order checked by reading, not run on hardware
//
#include <hip/hip_runtime.h>

// ---------------------------------------------------------------------------
// Mamba encoder for MI455X (gfx1250, wave32, WMMA + async-LDS data movement).
// Batched bf16 WMMA GEMMs over all B*T tokens + parallel depthwise conv +
// per-channel sequential SSM scan.
// ---------------------------------------------------------------------------

#define L_  4
#define DM_ 1024
#define DI_ 2048
#define DS_ 16
#define DC_ 4
#define DR_ 64
#define DF_ 4096
#define B_  8
#define T_  1024
#define M_  (B_ * T_)   // 8192 token rows

typedef __bf16 bf16;
typedef __attribute__((ext_vector_type(16))) __bf16 v16bf;
typedef __attribute__((ext_vector_type(8)))  __bf16 bf16x8;
typedef __attribute__((ext_vector_type(8)))  float  v8f;

// ---------------------------------------------------------------------------
// Async memory->LDS 16B move (GLOBAL_LOAD_ASYNC_TO_LDS_B128, ASYNCcnt).
// dst VGPR carries the LDS byte address (low 32 bits of the generic pointer).
// ---------------------------------------------------------------------------
__device__ __forceinline__ void async_ld16(const bf16* g, const bf16* l)
{
    unsigned           lds = (unsigned)(unsigned long long)l;
    unsigned long long ga  = (unsigned long long)g;
    asm volatile("global_load_async_to_lds_b128 %0, %1, off"
                 :: "v"(lds), "v"(ga) : "memory");
}
__device__ __forceinline__ void wait_async0()
{
    asm volatile("s_wait_asynccnt 0x0" ::: "memory");
}

// ---------------------------------------------------------------------------
// bf16 GEMM: C[M,N] (f32) = act(A[M,K] @ W[N,K]^T + bias) + residual
// A row stride = lda (elements). W is row-major (N x K), K-contiguous.
// act: 0=none, 1=softplus, 2=gelu(exact). Optional bf16 shadow copy Cbf.
// Block tile 128x64, 8 waves (4x2), wave tile 32x32 -> 4 wmma 16x16x32 per
// 32-wide K step. Double-buffered LDS fed by async global->LDS DMA.
// ---------------------------------------------------------------------------
__global__ __launch_bounds__(256)
void gemm_bf16_wmma(const bf16* __restrict__ A, int lda,
                    const bf16* __restrict__ W,
                    float* __restrict__ C, bf16* __restrict__ Cbf,
                    const float* __restrict__ bias,
                    const float* __restrict__ residual,
                    int N, int K, int act)
{
    __shared__ bf16 As[2][128][32];   // 2 x 8KB
    __shared__ bf16 Ws[2][64][32];    // 2 x 4KB

    const int tid   = threadIdx.x;
    const int lane  = tid & 31;
    const int wave  = tid >> 5;     // 0..7
    const int waveM = wave >> 1;    // 0..3 -> 32-row slices
    const int waveN = wave & 1;     // 0..1 -> 32-col slices
    const int tileM = blockIdx.y * 128;
    const int tileN = blockIdx.x * 64;

    const int ldRow   = tid >> 2;        // 0..63
    const int ldChunk = (tid & 3) * 8;   // 0,8,16,24 (bf16 elements)

    const int half = lane >> 4;          // k-phase per ISA 16-bit layouts
    const int r    = lane & 15;

    v8f acc[2][2] = {};

    // Stage one K-tile: A rows ldRow and ldRow+64, W row ldRow (16B per lane).
    auto issue_tiles = [&](int k0, int p) {
        async_ld16(A + (size_t)(tileM + ldRow) * lda + k0 + ldChunk,
                   &As[p][ldRow][ldChunk]);
        async_ld16(A + (size_t)(tileM + 64 + ldRow) * lda + k0 + ldChunk,
                   &As[p][64 + ldRow][ldChunk]);
        async_ld16(W + (size_t)(tileN + ldRow) * K + k0 + ldChunk,
                   &Ws[p][ldRow][ldChunk]);
    };

    issue_tiles(0, 0);
    int p = 0;
    for (int k0 = 0; k0 < K; k0 += 32, p ^= 1) {
        wait_async0();        // this thread's tile-p DMA complete
        __syncthreads();      // all waves: tile p populated, prior compute done
        if (k0 + 32 < K) issue_tiles(k0 + 32, p ^ 1);

        // Fragments (ISA 16-bit layouts): lane<16 holds K {0..7,16..23},
        // lane>=16 K {8..15,24..31}; A rows / B(=W) rows indexed by lane&15.
        union U { v16bf v; bf16x8 h[2]; };
        U a0, a1, b0, b1;
        {
            const bf16* row = &As[p][waveM * 32 + r][0];
            a0.h[0] = *(const bf16x8*)(row + half * 8);
            a0.h[1] = *(const bf16x8*)(row + 16 + half * 8);
        }
        {
            const bf16* row = &As[p][waveM * 32 + 16 + r][0];
            a1.h[0] = *(const bf16x8*)(row + half * 8);
            a1.h[1] = *(const bf16x8*)(row + 16 + half * 8);
        }
        {
            const bf16* row = &Ws[p][waveN * 32 + r][0];
            b0.h[0] = *(const bf16x8*)(row + half * 8);
            b0.h[1] = *(const bf16x8*)(row + 16 + half * 8);
        }
        {
            const bf16* row = &Ws[p][waveN * 32 + 16 + r][0];
            b1.h[0] = *(const bf16x8*)(row + half * 8);
            b1.h[1] = *(const bf16x8*)(row + 16 + half * 8);
        }

        acc[0][0] = __builtin_amdgcn_wmma_f32_16x16x32_bf16(false, a0.v, false, b0.v,
                                                            (short)0, acc[0][0], false, false);
        acc[0][1] = __builtin_amdgcn_wmma_f32_16x16x32_bf16(false, a0.v, false, b1.v,
                                                            (short)0, acc[0][1], false, false);
        acc[1][0] = __builtin_amdgcn_wmma_f32_16x16x32_bf16(false, a1.v, false, b0.v,
                                                            (short)0, acc[1][0], false, false);
        acc[1][1] = __builtin_amdgcn_wmma_f32_16x16x32_bf16(false, a1.v, false, b1.v,
                                                            (short)0, acc[1][1], false, false);
    }

    // Epilogue. C/D layout: VGPR i -> M = base + (lane<16 ? i : i+8), N = lane&15.
    for (int mi = 0; mi < 2; ++mi) {
        for (int nj = 0; nj < 2; ++nj) {
            v8f a = acc[mi][nj];
            int col = tileN + waveN * 32 + nj * 16 + r;
            if (col >= N) continue;
            float bv = bias ? bias[col] : 0.f;
            #pragma unroll
            for (int i = 0; i < 8; ++i) {
                int row = tileM + waveM * 32 + mi * 16 + half * 8 + i;
                float v = a[i] + bv;
                if (act == 1) {                       // softplus
                    v = (v > 20.f) ? v : log1pf(__expf(v));
                } else if (act == 2) {                // exact gelu
                    v = 0.5f * v * (1.f + erff(v * 0.70710678118654752f));
                }
                size_t off = (size_t)row * N + col;
                if (residual) v += residual[off];
                C[off] = v;
                if (Cbf) Cbf[off] = (bf16)v;
            }
        }
    }
}

// ---------------------------------------------------------------------------
// LayerNorm over last dim (DM_), one block per token row, bf16 output.
// ---------------------------------------------------------------------------
__global__ __launch_bounds__(256)
void layernorm_bf16(const float* __restrict__ x, const float* __restrict__ w,
                    const float* __restrict__ b, bf16* __restrict__ out)
{
    __shared__ float red[256];
    const int row = blockIdx.x;
    const float* xr = x + (size_t)row * DM_;

    float s = 0.f;
    for (int i = threadIdx.x; i < DM_; i += 256) s += xr[i];
    red[threadIdx.x] = s; __syncthreads();
    for (int st = 128; st > 0; st >>= 1) {
        if (threadIdx.x < st) red[threadIdx.x] += red[threadIdx.x + st];
        __syncthreads();
    }
    float mu = red[0] * (1.f / DM_);
    __syncthreads();

    float v = 0.f;
    for (int i = threadIdx.x; i < DM_; i += 256) { float d = xr[i] - mu; v += d * d; }
    red[threadIdx.x] = v; __syncthreads();
    for (int st = 128; st > 0; st >>= 1) {
        if (threadIdx.x < st) red[threadIdx.x] += red[threadIdx.x + st];
        __syncthreads();
    }
    float rs = rsqrtf(red[0] * (1.f / DM_) + 1e-5f);

    for (int i = threadIdx.x; i < DM_; i += 256)
        out[(size_t)row * DM_ + i] = (bf16)((xr[i] - mu) * rs * w[i] + b[i]);
}

// ---------------------------------------------------------------------------
// Causal depthwise conv (window DC_=4) + SiLU, parallel over (b,t,d).
// xz is (B*T, 2*DI): xc in cols [0,DI), z in cols [DI,2DI).
// Threads at t==T-1 emit the final conv state (last 4 xc values).
// ---------------------------------------------------------------------------
__global__ __launch_bounds__(256)
void conv_silu(const float* __restrict__ xz,
               const float* __restrict__ cw, const float* __restrict__ cb,
               const float* __restrict__ cstate_in,
               float* __restrict__ xs_f, bf16* __restrict__ xs_bf,
               float* __restrict__ cstate_out)
{
    int idx = blockIdx.x * 256 + threadIdx.x;
    if (idx >= B_ * T_ * DI_) return;
    int d = idx % DI_;
    int t = (idx / DI_) % T_;
    int b = idx / (DI_ * T_);

    float acc = cb[d];
    float win[DC_];
    #pragma unroll
    for (int c = 0; c < DC_; ++c) {
        int tt = t - (DC_ - 1) + c;
        float xv;
        if (tt >= 0) xv = xz[(size_t)(b * T_ + tt) * (2 * DI_) + d];
        else         xv = cstate_in[((size_t)b * DI_ + d) * DC_ + (DC_ + tt)];
        win[c] = xv;
        acc += xv * cw[d * DC_ + c];
    }
    float s = acc / (1.f + __expf(-acc));
    size_t o = (size_t)(b * T_ + t) * DI_ + d;
    xs_f[o] = s;
    xs_bf[o] = (bf16)s;
    if (t == T_ - 1) {
        #pragma unroll
        for (int c = 0; c < DC_; ++c)
            cstate_out[((size_t)b * DI_ + d) * DC_ + c] = win[c];
    }
}

// ---------------------------------------------------------------------------
// SSM selective scan: one thread per (b,d) channel, DS_=16-wide state,
// sequential over T. Fuses y = ssm . Cs + D*xs, gate y*silu(z), bf16 out.
// ---------------------------------------------------------------------------
__global__ __launch_bounds__(256)
void ssm_scan(const float* __restrict__ dt,    // (BT,DI) softplus'ed
              const float* __restrict__ xs,    // (BT,DI)
              const float* __restrict__ xdb,   // (BT,96): [64,80)=Bs, [80,96)=Cs
              const float* __restrict__ xz,    // (BT,2DI): z at col DI+d
              const float* __restrict__ A_log, // (DI,DS)
              const float* __restrict__ Dp,    // (DI)
              const float* __restrict__ s_in,  // (B,DI,DS)
              bf16* __restrict__ yg_bf,        // (BT,DI)
              float* __restrict__ s_out)       // (B,DI,DS)
{
    int idx = blockIdx.x * 256 + threadIdx.x;
    if (idx >= B_ * DI_) return;
    int d = idx % DI_;
    int b = idx / DI_;

    float A[DS_], st[DS_];
    #pragma unroll
    for (int n = 0; n < DS_; ++n) {
        A[n]  = -__expf(A_log[d * DS_ + n]);
        st[n] = s_in[((size_t)b * DI_ + d) * DS_ + n];
    }
    const float Dv = Dp[d];

    for (int t = 0; t < T_; ++t) {
        size_t rt  = (size_t)(b * T_ + t);
        float dtv  = dt[rt * DI_ + d];
        float xv   = xs[rt * DI_ + d];
        float du   = dtv * xv;
        const float* xr = xdb + rt * (DR_ + 2 * DS_);
        float y = Dv * xv;
        #pragma unroll
        for (int n = 0; n < DS_; ++n) {
            st[n] = st[n] * __expf(dtv * A[n]) + du * xr[DR_ + n];
            y += st[n] * xr[DR_ + DS_ + n];
        }
        float z = xz[rt * (2 * DI_) + DI_ + d];
        float g = z / (1.f + __expf(-z));
        yg_bf[rt * DI_ + d] = (bf16)(y * g);
    }
    #pragma unroll
    for (int n = 0; n < DS_; ++n)
        s_out[((size_t)b * DI_ + d) * DS_ + n] = st[n];
}

// ---------------------------------------------------------------------------
// Elementwise helpers
// ---------------------------------------------------------------------------
__global__ void cvt_f32_bf16(const float* __restrict__ s, bf16* __restrict__ d, int n)
{
    int i = blockIdx.x * 256 + threadIdx.x;
    if (i < n) d[i] = (bf16)s[i];
}
__global__ void copy_f32(const float* __restrict__ s, float* __restrict__ d, int n)
{
    int i = blockIdx.x * 256 + threadIdx.x;
    if (i < n) d[i] = s[i];
}

// ---------------------------------------------------------------------------
// Host orchestration
// ---------------------------------------------------------------------------
extern "C" void kernel_launch(void* const* d_in, const int* in_sizes, int n_in,
                              void* d_out, int out_size, void* d_ws, size_t ws_size,
                              hipStream_t stream)
{
    const float* in_x      = (const float*)d_in[0];
    const float* in_cstate = (const float*)d_in[2];
    const float* in_sstate = (const float*)d_in[3];
    const float* ln1_w = (const float*)d_in[4];
    const float* ln1_b = (const float*)d_in[5];
    const float* ln2_w = (const float*)d_in[6];
    const float* ln2_b = (const float*)d_in[7];
    const float* in_proj_w = (const float*)d_in[8];
    const float* conv_w    = (const float*)d_in[9];
    const float* conv_b    = (const float*)d_in[10];
    const float* x_proj_w  = (const float*)d_in[11];
    const float* dt_proj_w = (const float*)d_in[12];
    const float* dt_proj_b = (const float*)d_in[13];
    const float* A_log     = (const float*)d_in[14];
    const float* D_param   = (const float*)d_in[15];
    const float* out_proj_w = (const float*)d_in[16];
    const float* ffn_w1 = (const float*)d_in[17];
    const float* ffn_b1 = (const float*)d_in[18];
    const float* ffn_w2 = (const float*)d_in[19];
    const float* ffn_b2 = (const float*)d_in[20];

    float* out_x    = (float*)d_out;
    float* out_conv = out_x + (size_t)M_ * DM_;
    float* out_ssm  = out_conv + (size_t)L_ * B_ * DI_ * DC_;

    // Workspace carve-up
    char* ws = (char*)d_ws;
    size_t off = 0;
    auto alloc = [&](size_t bytes) -> void* {
        void* p = ws + off;
        off += (bytes + 255) & ~(size_t)255;
        return p;
    };
    float* x_cur  = (float*)alloc((size_t)M_ * DM_ * 4);
    bf16*  xn_bf  = (bf16*) alloc((size_t)M_ * DM_ * 2);        // LN1 & LN2 output
    float* xz     = (float*)alloc((size_t)M_ * 2 * DI_ * 4);    // also reused as FFN hidden
    float* xs_f   = (float*)alloc((size_t)M_ * DI_ * 4);
    bf16*  xs_bf  = (bf16*) alloc((size_t)M_ * DI_ * 2);
    float* xdb_f  = (float*)alloc((size_t)M_ * (DR_ + 2 * DS_) * 4);
    bf16*  xdb_bf = (bf16*) alloc((size_t)M_ * (DR_ + 2 * DS_) * 2);
    float* dt_f   = (float*)alloc((size_t)M_ * DI_ * 4);
    bf16*  yg_bf  = (bf16*) alloc((size_t)M_ * DI_ * 2);
    bf16*  h_bf   = (bf16*) alloc((size_t)M_ * DF_ * 2);
    bf16*  w_in   = (bf16*) alloc((size_t)2 * DI_ * DM_ * 2);
    bf16*  w_xp   = (bf16*) alloc((size_t)(DR_ + 2 * DS_) * DI_ * 2);
    bf16*  w_dtp  = (bf16*) alloc((size_t)DI_ * DR_ * 2);
    bf16*  w_out  = (bf16*) alloc((size_t)DM_ * DI_ * 2);
    bf16*  w_f1   = (bf16*) alloc((size_t)DF_ * DM_ * 2);
    bf16*  w_f2   = (bf16*) alloc((size_t)DM_ * DF_ * 2);

    const dim3 blk(256);
    auto g1 = [](int n) { return dim3((n + 255) / 256); };
    auto gemm_grid = [](int N) { return dim3((N + 63) / 64, M_ / 128); };

    // residual stream init
    copy_f32<<<g1(M_ * DM_), blk, 0, stream>>>(in_x, x_cur, M_ * DM_);

    for (int l = 0; l < L_; ++l) {
        // --- per-layer weight views + bf16 conversion -----------------------
        const float* inw = in_proj_w + (size_t)l * 2 * DI_ * DM_;
        const float* xpw = x_proj_w  + (size_t)l * (DR_ + 2 * DS_) * DI_;
        const float* dtw = dt_proj_w + (size_t)l * DI_ * DR_;
        const float* ow  = out_proj_w + (size_t)l * DM_ * DI_;
        const float* f1w = ffn_w1 + (size_t)l * DF_ * DM_;
        const float* f2w = ffn_w2 + (size_t)l * DM_ * DF_;
        cvt_f32_bf16<<<g1(2 * DI_ * DM_), blk, 0, stream>>>(inw, w_in, 2 * DI_ * DM_);
        cvt_f32_bf16<<<g1((DR_ + 2 * DS_) * DI_), blk, 0, stream>>>(xpw, w_xp, (DR_ + 2 * DS_) * DI_);
        cvt_f32_bf16<<<g1(DI_ * DR_), blk, 0, stream>>>(dtw, w_dtp, DI_ * DR_);
        cvt_f32_bf16<<<g1(DM_ * DI_), blk, 0, stream>>>(ow, w_out, DM_ * DI_);
        cvt_f32_bf16<<<g1(DF_ * DM_), blk, 0, stream>>>(f1w, w_f1, DF_ * DM_);
        cvt_f32_bf16<<<g1(DM_ * DF_), blk, 0, stream>>>(f2w, w_f2, DM_ * DF_);

        // --- LN1 -> in_proj GEMM (M x 2DI) ---------------------------------
        layernorm_bf16<<<dim3(M_), blk, 0, stream>>>(
            x_cur, ln1_w + (size_t)l * DM_, ln1_b + (size_t)l * DM_, xn_bf);
        gemm_bf16_wmma<<<gemm_grid(2 * DI_), blk, 0, stream>>>(
            xn_bf, DM_, w_in, xz, nullptr, nullptr, nullptr, 2 * DI_, DM_, 0);

        // --- depthwise conv + SiLU -----------------------------------------
        conv_silu<<<g1(B_ * T_ * DI_), blk, 0, stream>>>(
            xz, conv_w + (size_t)l * DI_ * DC_, conv_b + (size_t)l * DI_,
            in_cstate + (size_t)l * B_ * DI_ * DC_,
            xs_f, xs_bf, out_conv + (size_t)l * B_ * DI_ * DC_);

        // --- x_proj GEMM (M x 96) ------------------------------------------
        gemm_bf16_wmma<<<gemm_grid(DR_ + 2 * DS_), blk, 0, stream>>>(
            xs_bf, DI_, w_xp, xdb_f, xdb_bf, nullptr, nullptr, DR_ + 2 * DS_, DI_, 0);

        // --- dt_proj GEMM + softplus (M x DI, K=DR) ------------------------
        gemm_bf16_wmma<<<gemm_grid(DI_), blk, 0, stream>>>(
            xdb_bf, DR_ + 2 * DS_, w_dtp, dt_f, nullptr,
            dt_proj_b + (size_t)l * DI_, nullptr, DI_, DR_, 1);

        // --- selective scan + gating ---------------------------------------
        ssm_scan<<<g1(B_ * DI_), blk, 0, stream>>>(
            dt_f, xs_f, xdb_f, xz,
            A_log + (size_t)l * DI_ * DS_, D_param + (size_t)l * DI_,
            in_sstate + (size_t)l * B_ * DI_ * DS_,
            yg_bf, out_ssm + (size_t)l * B_ * DI_ * DS_);

        // --- out_proj GEMM + residual add into x_cur -----------------------
        gemm_bf16_wmma<<<gemm_grid(DM_), blk, 0, stream>>>(
            yg_bf, DI_, w_out, x_cur, nullptr, nullptr, x_cur, DM_, DI_, 0);

        // --- FFN: LN2 -> W1+GELU -> W2 + residual --------------------------
        layernorm_bf16<<<dim3(M_), blk, 0, stream>>>(
            x_cur, ln2_w + (size_t)l * DM_, ln2_b + (size_t)l * DM_, xn_bf);
        gemm_bf16_wmma<<<gemm_grid(DF_), blk, 0, stream>>>(
            xn_bf, DM_, w_f1, xz /* reuse as h */, h_bf,
            ffn_b1 + (size_t)l * DF_, nullptr, DF_, DM_, 2);
        gemm_bf16_wmma<<<gemm_grid(DM_), blk, 0, stream>>>(
            h_bf, DF_, w_f2, x_cur, nullptr,
            ffn_b2 + (size_t)l * DM_, x_cur, DM_, DF_, 0);
    }

    // final residual stream -> output
    copy_f32<<<g1(M_ * DM_), blk, 0, stream>>>(x_cur, out_x, M_ * DM_);
}
